// LSTM_Cell_80891414053127
// MI455X (gfx1250) — compile-verified, matrix operand order checked by reading, not run on hardware
//
#include <hip/hip_runtime.h>
#include <math.h>

// LSTM cell, B=4096, D=H=1024. Fused single GEMM [4096x2048]x[2048x4096]
// using V_WMMA_F32_16X16X32_F16 (f16 hi/lo split, 3 products, fp32 accum)
// with Tensor Data Mover (tensor_load_to_lds) staging global -> LDS.
// Requires ws_size >= 64 MB.

#define Bm   4096
#define Hh   1024
#define KTOT 2048
#define LDSS 40               // LDS row stride in halves (pad: +4 dwords per 16 dwords)
#define TILE (128 * LDSS)     // one 128x32 tile in halves (10240 B)
#define SGS  33               // epilogue LDS stride (floats)
#define PLANE (128 * SGS)

typedef __attribute__((ext_vector_type(16))) _Float16 v16h;
typedef __attribute__((ext_vector_type(8)))  float    v8f;
typedef __attribute__((ext_vector_type(4)))  unsigned u32x4;
typedef __attribute__((ext_vector_type(8)))  int      i32x8;
typedef __attribute__((ext_vector_type(4)))  int      i32x4;

// ---- pass 1: fp32 -> f16 hi/lo, rows are [src0 row (1024) | src1 row (1024)] ----
__global__ __launch_bounds__(256) void split_rows_f16(
    const float* __restrict__ s0, const float* __restrict__ s1,
    _Float16* __restrict__ hi, _Float16* __restrict__ lo) {
  size_t idx = (size_t)blockIdx.x * 256 + threadIdx.x;   // < 4096*2048
  size_t row = idx >> 11;
  int    k   = (int)(idx & 2047);
  float  v   = (k < 1024) ? s0[row * 1024 + k] : s1[row * 1024 + (k - 1024)];
  _Float16 h = (_Float16)v;
  hi[idx] = h;
  lo[idx] = (_Float16)(v - (float)h);
}

// ---- TDM: DMA one strided f16 tile (32 elems wide) into LDS with HW padding ----
// tile_d1 = rows per z-block, tile_d2 = z blocks (0 => 2D), s1 = z stride (elems)
__device__ __forceinline__ void tdm_issue(unsigned lds_addr, const _Float16* gptr,
                                          unsigned tile_d1, unsigned tile_d2,
                                          unsigned long long s1) {
  unsigned long long ga = (unsigned long long)(uintptr_t)gptr;
  const unsigned D0 = 2048, D1 = 4096;           // tensor extents (no clipping)
  const unsigned long long S0 = 2048;            // row stride in elements

  u32x4 g0;
  g0[0] = 1u;                                    // count=1 valid, user mode
  g0[1] = lds_addr;                              // LDS byte address
  g0[2] = (unsigned)ga;                          // global_addr[31:0]
  g0[3] = (unsigned)((ga >> 32) & 0x1FFFFFFu) | (2u << 30);  // ga[56:32] | type=2

  i32x8 g1;
  g1[0] = (int)((1u << 16)      // data_size = 2 bytes
              | (1u << 20)      // pad_enable
              | (3u << 22)      // pad_interval: 16 DWORDs (32 halves)
              | (3u << 25));    // pad_amount: 4 DWORDs (8 halves) -> stride 40
  g1[1] = (int)((D0 & 0xFFFFu) << 16);                    // tensor_dim0[15:0]
  g1[2] = (int)((D0 >> 16) | ((D1 & 0xFFFFu) << 16));     // dim0[31:16] | dim1[15:0]
  g1[3] = (int)((D1 >> 16) | (32u << 16));                // dim1[31:16] | tile_dim0=32
  g1[4] = (int)(tile_d1 | (tile_d2 << 16));               // tile_dim1 | tile_dim2
  g1[5] = (int)(unsigned)S0;                              // dim0_stride[31:0]
  g1[6] = (int)((unsigned)(S0 >> 32) | ((unsigned)(s1 & 0xFFFFull) << 16));
  g1[7] = (int)(unsigned)(s1 >> 16);                      // dim1_stride[47:16]

  i32x4 g2;
  g2[0] = 4; g2[1] = 0; g2[2] = 0; g2[3] = 0;             // tensor_dim2=4 (z OOB)
  i32x4 g3;
  g3[0] = 0; g3[1] = 0; g3[2] = 0; g3[3] = 0;

#if defined(__clang_major__) && __clang_major__ >= 23
  i32x8 g4;
  g4[0]=0; g4[1]=0; g4[2]=0; g4[3]=0; g4[4]=0; g4[5]=0; g4[6]=0; g4[7]=0;
  __builtin_amdgcn_tensor_load_to_lds(g0, g1, g2, g3, g4, 0);
#else
  __builtin_amdgcn_tensor_load_to_lds(g0, g1, g2, g3, 0);
#endif
}

// ---- pass 2: fused GEMM + LSTM epilogue ----
__global__ __launch_bounds__(256) void lstm_wmma(
    const _Float16* __restrict__ Ahi, const _Float16* __restrict__ Alo,
    const _Float16* __restrict__ Bhi, const _Float16* __restrict__ Blo,
    const float* __restrict__ bW, const float* __restrict__ bU,
    const float* __restrict__ cprev, float* __restrict__ out) {
  extern __shared__ __align__(16) char smem[];
  _Float16* s = (_Float16*)smem;

  const int tid   = threadIdx.x;
  const int lane  = tid & 31;
  const int w     = tid >> 5;     // wave 0..7
  const int wm    = w & 1;        // M half of block tile
  const int wg    = w >> 1;       // gate 0..3
  const int mbase = blockIdx.y * 128;
  const int hbase = blockIdx.x * 32;

  // low 32 bits of the flat LDS pointer are the LDS byte address
  const unsigned ldsBase = (unsigned)(uintptr_t)(void*)smem;

  v8f acc[4][2] = {};

  // issue all 4 tile DMAs for one k-step into buffer `buf` (wave 0 only)
  auto issue = [&](int buf, int k0) {
    unsigned lb = ldsBase + (unsigned)(buf * 4) * (TILE * 2);
    size_t ka = (size_t)mbase * KTOT + k0;
    size_t kb = (size_t)hbase * KTOT + k0;
    tdm_issue(lb + 0 * TILE * 2, Ahi + ka, 128, 0, 0);                    // A hi (2D)
    tdm_issue(lb + 1 * TILE * 2, Alo + ka, 128, 0, 0);                    // A lo (2D)
    tdm_issue(lb + 2 * TILE * 2, Bhi + kb, 32, 4, 1024ull * KTOT);        // B hi (3D, z=gate)
    tdm_issue(lb + 3 * TILE * 2, Blo + kb, 32, 4, 1024ull * KTOT);        // B lo (3D)
  };

  // 16-bit WMMA fragment: lanes 0-15 take K{0..7,16..23}, lanes 16-31 K{8..15,24..31}
  auto frag = [&](const _Float16* t, int r0) -> v16h {
    const _Float16* p = t + (size_t)(r0 + (lane & 15)) * LDSS + (lane >> 4) * 8;
    union { v16h v; uint4 q[2]; } u;
    u.q[0] = *(const uint4*)(p);
    u.q[1] = *(const uint4*)(p + 16);
    return u.v;
  };

  if (w == 0) {
    issue(0, 0);
    __builtin_amdgcn_s_wait_tensorcnt(0);
  }
  __syncthreads();

  const int NK = KTOT / 32;
  for (int kt = 0; kt < NK; ++kt) {
    int cur = kt & 1;
    if (w == 0 && kt + 1 < NK) issue(cur ^ 1, (kt + 1) * 32);

    const _Float16* tAh = s + (cur * 4 + 0) * TILE;
    const _Float16* tAl = s + (cur * 4 + 1) * TILE;
    const _Float16* tBh = s + (cur * 4 + 2) * TILE;
    const _Float16* tBl = s + (cur * 4 + 3) * TILE;

    v16h ah[4], al[4], bh[2], bl[2];
#pragma unroll
    for (int i = 0; i < 4; ++i) {
      ah[i] = frag(tAh, wm * 64 + i * 16);
      al[i] = frag(tAl, wm * 64 + i * 16);
    }
#pragma unroll
    for (int j = 0; j < 2; ++j) {
      bh[j] = frag(tBh, wg * 32 + j * 16);
      bl[j] = frag(tBl, wg * 32 + j * 16);
    }
#pragma unroll
    for (int i = 0; i < 4; ++i)
#pragma unroll
      for (int j = 0; j < 2; ++j) {
        acc[i][j] = __builtin_amdgcn_wmma_f32_16x16x32_f16(
            false, ah[i], false, bh[j], (short)0, acc[i][j], false, false);
        acc[i][j] = __builtin_amdgcn_wmma_f32_16x16x32_f16(
            false, ah[i], false, bl[j], (short)0, acc[i][j], false, false);
        acc[i][j] = __builtin_amdgcn_wmma_f32_16x16x32_f16(
            false, al[i], false, bh[j], (short)0, acc[i][j], false, false);
      }

    if (w == 0 && kt + 1 < NK) __builtin_amdgcn_s_wait_tensorcnt(0);
    __syncthreads();
  }

  // ---- epilogue: exchange 4 gate pre-activations through LDS (reuse staging) ----
  float* sG = (float*)smem;
#pragma unroll
  for (int j = 0; j < 2; ++j) {
    int hcol = j * 16 + (lane & 15);
    float bias = bW[wg * Hh + hbase + hcol] + bU[wg * Hh + hbase + hcol];
#pragma unroll
    for (int i = 0; i < 4; ++i) {
      int mrow = wm * 64 + i * 16 + ((lane >> 4) << 3);  // C layout: VGPR r -> M=r / M=r+8
#pragma unroll
      for (int r = 0; r < 8; ++r)
        sG[wg * PLANE + (mrow + r) * SGS + hcol] = acc[i][j][r] + bias;
    }
  }
  __syncthreads();

  float* outh = out;
  float* outc = out + (size_t)Bm * Hh;
  for (int e = tid; e < 128 * 32; e += 256) {
    int m  = e >> 5;
    int hh = e & 31;
    float gi = sG[0 * PLANE + m * SGS + hh];
    float gf = sG[1 * PLANE + m * SGS + hh];
    float go = sG[2 * PLANE + m * SGS + hh];
    float gz = sG[3 * PLANE + m * SGS + hh];
    float ii = 1.0f / (1.0f + expf(-gi));
    float ff = 1.0f / (1.0f + expf(-gf));
    float oo = 1.0f / (1.0f + expf(-go));
    float zz = tanhf(gz);
    size_t gidx = (size_t)(mbase + m) * Hh + hbase + hh;
    float ct = ii * zz + ff * cprev[gidx];
    outh[gidx] = oo * tanhf(ct);
    outc[gidx] = ct;
  }
}

extern "C" void kernel_launch(void* const* d_in, const int* in_sizes, int n_in,
                              void* d_out, int out_size, void* d_ws, size_t ws_size,
                              hipStream_t stream) {
  (void)in_sizes; (void)n_in; (void)out_size; (void)ws_size;
  const float* x_t   = (const float*)d_in[0];
  const float* h_t_1 = (const float*)d_in[1];
  const float* c_t_1 = (const float*)d_in[2];
  const float* W     = (const float*)d_in[3];
  const float* bW    = (const float*)d_in[4];
  const float* U     = (const float*)d_in[5];
  const float* bU    = (const float*)d_in[6];

  const size_t SEG = (size_t)Bm * KTOT * sizeof(_Float16);  // 16 MB
  _Float16* Ahi = (_Float16*)((char*)d_ws + 0 * SEG);
  _Float16* Alo = (_Float16*)((char*)d_ws + 1 * SEG);
  _Float16* Bhi = (_Float16*)((char*)d_ws + 2 * SEG);
  _Float16* Blo = (_Float16*)((char*)d_ws + 3 * SEG);

  const int nElem  = Bm * KTOT;          // 8,388,608
  const int nBlk   = nElem / 256;        // 32768
  split_rows_f16<<<nBlk, 256, 0, stream>>>(x_t, h_t_1, Ahi, Alo);   // A = [x | h]
  split_rows_f16<<<nBlk, 256, 0, stream>>>(W,   U,     Bhi, Blo);   // B^T = [W | U]

  const size_t smem = 8 * TILE * sizeof(_Float16);  // 81920 B staging (> epilogue 67584 B)
  lstm_wmma<<<dim3(32, 32), 256, smem, stream>>>(Ahi, Alo, Bhi, Blo, bW, bU, c_t_1,
                                                 (float*)d_out);
}
